// Darkfarseer_80350248173889
// MI455X (gfx1250) — matrix operand order repaired and checked by reading, not compiled, not run on hardware
//
#include <hip/hip_runtime.h>
#include <math.h>

// ---------------- problem constants ----------------
#define NN     5000          // nodes
#define NPAD   5024          // K padded to multiple of 32 (157*32)
#define MPAD   5120          // padded row count for S matrices
#define NGRP   97            // 96 (b,t) groups + 1 aug group
#define NCOL   (NGRP*64)     // 6208 columns of H1
#define BT     96
#define TT     24
#define BB     4
#define DD     64
#define MMIS   256
#define KNEG   20
#define MASKT  4
// LDS row strides (halfwords), 16B-aligned rows
#define AS     40
#define BSL    40

typedef __attribute__((ext_vector_type(16))) __bf16 v16bf;
typedef __attribute__((ext_vector_type(8)))  float  v8f;

struct BF8x2 { uint4 a, b; };
static __device__ __forceinline__ v16bf ld2(const __bf16* p0, const __bf16* p1) {
  BF8x2 u; u.a = *(const uint4*)p0; u.b = *(const uint4*)p1;
  return __builtin_bit_cast(v16bf, u);
}

// async 16-byte copy global -> LDS (CDNA5 VGLOBAL async, tracked by ASYNCcnt)
static __device__ __forceinline__ void async_cp16(const __bf16* gsrc, unsigned lds_off) {
  asm volatile("global_load_async_to_lds_b128 %0, %1, off"
               :: "v"(lds_off), "v"((unsigned long long)(uintptr_t)gsrc)
               : "memory");
}

// ---------------- K0: layer-1 coefficient vectors ----------------
__global__ void k_coef(const float* __restrict__ Ws, const float* __restrict__ bs,
                       const float* __restrict__ W1, const float* __restrict__ b1,
                       float* __restrict__ coef) {
  int d = threadIdx.x;
  if (d < 64) {
    float a1=0,a2=0,a3=0,s0=0,s1=0,s2=0;
    for (int c=0;c<64;++c) {
      float w = Ws[c], b = bs[c];
      a1 = fmaf(w, W1[c*64+d], a1);
      a2 = fmaf(w, W1[(64+c)*64+d], a2);
      a3 = fmaf(w, W1[(128+c)*64+d], a3);
      s0 = fmaf(b, W1[c*64+d], s0);
      s1 = fmaf(b, W1[(64+c)*64+d], s1);
      s2 = fmaf(b, W1[(128+c)*64+d], s2);
    }
    coef[0*64+d]=a1; coef[1*64+d]=a2; coef[2*64+d]=a3;
    coef[3*64+d]=s0 + b1[d]; coef[4*64+d]=s1; coef[5*64+d]=s2;
  }
}

// ---------------- K1: degrees / inverse / rowsum-of-S ----------------
__global__ void k_rowdeg(const float* __restrict__ adj, float* __restrict__ dinvr,
                         float* __restrict__ rs1) {
  int m = blockIdx.x;
  __shared__ float red[256];
  float s = 0.f;
  for (int n = threadIdx.x; n < NN; n += 256) s += adj[(size_t)m*NN + n];
  red[threadIdx.x] = s; __syncthreads();
  for (int o = 128; o > 0; o >>= 1) {
    if (threadIdx.x < o) red[threadIdx.x] += red[threadIdx.x + o];
    __syncthreads();
  }
  if (threadIdx.x == 0) {
    float deg = red[0];
    float di = deg > 0.f ? 1.f/deg : 0.f;
    dinvr[m] = di; rs1[m] = di*deg;
  }
}

__global__ void k_coldeg(const float* __restrict__ adj, float* __restrict__ dinvc,
                         float* __restrict__ rs2) {
  int n = blockIdx.x*256 + threadIdx.x;
  if (n >= NN) return;
  float s = 0.f;
  for (int m = 0; m < NN; ++m) s += adj[(size_t)m*NN + n];
  float di = s > 0.f ? 1.f/s : 0.f;
  dinvc[n] = di; rs2[n] = di*s;
}

// ---------------- K2: x reshape + y1/y2 = S1@x, S2@x ----------------
__global__ void k_xr(const float* __restrict__ x, float* __restrict__ xr) {
  int i = blockIdx.x*256 + threadIdx.x;
  if (i >= BT*NN) return;
  int g = i / NN, n = i % NN;
  xr[(size_t)n*BT + g] = x[(size_t)g*NN + n];
}

__global__ void k_y12(const float* __restrict__ adj, const float* __restrict__ xr,
                      const float* __restrict__ dinvr, const float* __restrict__ dinvc,
                      float* __restrict__ y1, float* __restrict__ y2) {
  int m = blockIdx.x;
  int c = threadIdx.x;            // 96 threads
  float a1 = 0.f, a2 = 0.f;
  for (int n = 0; n < NN; ++n) {
    float xv = xr[(size_t)n*BT + c];
    a1 = fmaf(adj[(size_t)m*NN + n], xv, a1);
    a2 = fmaf(adj[(size_t)n*NN + m], xv, a2);
  }
  y1[(size_t)m*BT + c] = dinvr[m]*a1;
  y2[(size_t)m*BT + c] = dinvc[m]*a2;
}

// ---------------- K3: build H1 (relu layer-1 output) as bf16 hi/lo ----------------
__global__ void k_h1(const float* __restrict__ xr, const float* __restrict__ y1,
                     const float* __restrict__ y2, const float* __restrict__ rs1,
                     const float* __restrict__ rs2, const float* __restrict__ coef,
                     __bf16* __restrict__ Hhi, __bf16* __restrict__ Hlo) {
  __shared__ float sc[6*64];
  for (int i = threadIdx.x; i < 384; i += 256) sc[i] = coef[i];
  __syncthreads();
  int g = blockIdx.y;                         // 0..96
  int n = blockIdx.x*256 + threadIdx.x;       // 0..NPAD
  if (n >= NPAD) return;
  bool valid = n < NN;
  float xv = 0.f, y1v = 0.f, y2v = 0.f, r1 = 0.f, r2 = 0.f;
  if (valid) {
    r1 = rs1[n]; r2 = rs2[n];
    if (g < BT) { xv = xr[(size_t)n*BT+g]; y1v = y1[(size_t)n*BT+g]; y2v = y2[(size_t)n*BT+g]; }
  }
  for (int d = 0; d < 64; ++d) {
    float f = 0.f;
    if (valid) {
      f = sc[192+d] + r1*sc[256+d] + r2*sc[320+d];
      if (g < BT) f += sc[d]*xv + sc[64+d]*y1v + sc[128+d]*y2v;
      f = fmaxf(f, 0.f);
    }
    __bf16 hi = (__bf16)f;
    __bf16 lo = (__bf16)(f - (float)hi);
    size_t j = (size_t)(g*64 + d)*NPAD + n;
    Hhi[j] = hi; Hlo[j] = lo;
  }
}

// ---------------- K3b: S1 = dinvr*adj as padded bf16 hi/lo [MPAD][NPAD] ----------------
__global__ void k_s1conv(const float* __restrict__ adj, const float* __restrict__ dinvr,
                         __bf16* __restrict__ Shi, __bf16* __restrict__ Slo) {
  int m = blockIdx.y;
  int k = blockIdx.x*256 + threadIdx.x;
  if (k >= NPAD) return;
  float v = 0.f;
  if (m < NN && k < NN) v = adj[(size_t)m*NN + k] * dinvr[m];
  __bf16 hi = (__bf16)v;
  size_t o = (size_t)m*NPAD + k;
  Shi[o] = hi; Slo[o] = (__bf16)(v - (float)hi);
}

// ---------------- K3c: S2 = dinvc*adj^T via LDS-tiled transpose ----------------
__global__ __launch_bounds__(256)
void k_s2conv(const float* __restrict__ adj, const float* __restrict__ dinvc,
              __bf16* __restrict__ Shi, __bf16* __restrict__ Slo) {
  __shared__ float sT[64][65];
  const int k0 = blockIdx.x*64;     // column of S2 (row of adj)
  const int m0 = blockIdx.y*64;     // row of S2
  // read adj[k][m] coalesced over m, scale by dinvc[m]
  for (int idx = threadIdx.x; idx < 64*64; idx += 256) {
    int j = idx & 63, i = idx >> 6;       // j: m dir, i: k dir
    float v = 0.f;
    if (k0 + i < NN && m0 + j < NN) v = adj[(size_t)(k0+i)*NN + (m0+j)] * dinvc[m0+j];
    sT[i][j] = v;
  }
  __syncthreads();
  // write S2[m][k] coalesced over k
  for (int idx = threadIdx.x; idx < 64*64; idx += 256) {
    int ii = idx & 63, jj = idx >> 6;     // ii: k dir, jj: m dir
    int k = k0 + ii, m = m0 + jj;
    if (k < NPAD) {
      float v = sT[ii][jj];
      __bf16 hi = (__bf16)v;
      size_t o = (size_t)m*NPAD + k;
      Shi[o] = hi; Slo[o] = (__bf16)(v - (float)hi);
    }
  }
}

// ---------------- K4: O = S @ H1, split-bf16 WMMA, async double-buffered LDS ----------------
// 128 threads (4 waves). Block tile: 128 m x 64 j. Wave tile: 32 m x 64 j.
__global__ __launch_bounds__(128)
void k_mm(const __bf16* __restrict__ Shi, const __bf16* __restrict__ Slo,
          const __bf16* __restrict__ Hhi, const __bf16* __restrict__ Hlo,
          float* __restrict__ O) {
  __shared__ alignas(16) __bf16 sAhi[2][128*AS];
  __shared__ alignas(16) __bf16 sAlo[2][128*AS];
  __shared__ alignas(16) __bf16 sBhi[2][64*BSL];
  __shared__ alignas(16) __bf16 sBlo[2][64*BSL];

  const int j0 = blockIdx.x*64;
  const int m0 = blockIdx.y*128;
  const int tid = threadIdx.x;
  const int wave = tid >> 5, lane = tid & 31;
  const int lh = lane >> 4, l16 = lane & 15;

  // stage tile (k0) into buffer buf: 12 async b128 issues per wave
  auto stage = [&](int buf, int k0) {
    #pragma unroll
    for (int it = 0; it < 4; ++it) {            // A: 512 chunks of 16B
      int idx = it*128 + tid;
      int ch = idx & 3, mm = idx >> 2;
      size_t src = (size_t)(m0 + mm)*NPAD + k0 + ch*8;
      async_cp16(Shi + src, (unsigned)(uintptr_t)&sAhi[buf][mm*AS + ch*8]);
      async_cp16(Slo + src, (unsigned)(uintptr_t)&sAlo[buf][mm*AS + ch*8]);
    }
    #pragma unroll
    for (int it = 0; it < 2; ++it) {            // B: 256 chunks of 16B
      int idx = it*128 + tid;
      int ch = idx & 3, jj = idx >> 2;
      size_t src = (size_t)(j0 + jj)*NPAD + k0 + ch*8;
      async_cp16(Hhi + src, (unsigned)(uintptr_t)&sBhi[buf][jj*BSL + ch*8]);
      async_cp16(Hlo + src, (unsigned)(uintptr_t)&sBlo[buf][jj*BSL + ch*8]);
    }
  };

  v8f acc[2][4];
  #pragma unroll
  for (int a = 0; a < 2; ++a)
    #pragma unroll
    for (int b = 0; b < 4; ++b) acc[a][b] = (v8f){0.f,0.f,0.f,0.f,0.f,0.f,0.f,0.f};

  stage(0, 0);
  const int nk = NPAD/32;                       // 157
  for (int i = 0; i < nk; ++i) {
    const int buf = i & 1;
    __syncthreads();                            // all waves done with buffer buf^1
    if (i + 1 < nk) {
      stage(buf ^ 1, (i + 1)*32);
      asm volatile("s_wait_asynccnt 12" ::: "memory");  // tile i complete (in-order)
    } else {
      asm volatile("s_wait_asynccnt 0" ::: "memory");
    }
    __syncthreads();                            // tile i visible to all waves

    // A fragments (16x32 bf16): lane<16 -> K{0..7,16..23}, lane>=16 -> K{8..15,24..31}
    v16bf ahi[2], alo[2];
    #pragma unroll
    for (int mt = 0; mt < 2; ++mt) {
      const __bf16* ap = &sAhi[buf][(wave*32 + mt*16 + l16)*AS + lh*8];
      ahi[mt] = ld2(ap, ap + 16);
      const __bf16* aq = &sAlo[buf][(wave*32 + mt*16 + l16)*AS + lh*8];
      alo[mt] = ld2(aq, aq + 16);
    }
    #pragma unroll
    for (int jt = 0; jt < 4; ++jt) {
      // B fragment (32x16 bf16): lane<16 -> K{0..7,16..23} of col l16; lane>=16 -> K{8..15,24..31}
      const __bf16* bp = &sBhi[buf][(jt*16 + l16)*BSL + lh*8];
      v16bf bhi = ld2(bp, bp + 16);
      const __bf16* bq = &sBlo[buf][(jt*16 + l16)*BSL + lh*8];
      v16bf blo = ld2(bq, bq + 16);
      #pragma unroll
      for (int mt = 0; mt < 2; ++mt) {
        acc[mt][jt] = __builtin_amdgcn_wmma_f32_16x16x32_bf16(
            false, ahi[mt], false, bhi, (short)0, acc[mt][jt], false, false);
        acc[mt][jt] = __builtin_amdgcn_wmma_f32_16x16x32_bf16(
            false, ahi[mt], false, blo, (short)0, acc[mt][jt], false, false);
        acc[mt][jt] = __builtin_amdgcn_wmma_f32_16x16x32_bf16(
            false, alo[mt], false, bhi, (short)0, acc[mt][jt], false, false);
      }
    }
  }

  // store C fragments: VGPR r holds M = r + 8*(lane>=16), N = lane%16
  #pragma unroll
  for (int mt = 0; mt < 2; ++mt)
    #pragma unroll
    for (int jt = 0; jt < 4; ++jt) {
      int jcol = j0 + jt*16 + l16;
      #pragma unroll
      for (int r = 0; r < 8; ++r) {
        int m = m0 + wave*32 + mt*16 + lh*8 + r;
        if (m < NN) O[(size_t)m*NCOL + jcol] = acc[mt][jt][r];
      }
    }
}

// ---------------- K5: layer-2 channel combine (192 -> 64 with W2) ----------------
__global__ __launch_bounds__(256)
void k_combine(const float* __restrict__ O1, const float* __restrict__ O2,
               const __bf16* __restrict__ Hhi, const __bf16* __restrict__ Hlo,
               const float* __restrict__ W2, const float* __restrict__ b2,
               float* __restrict__ h2, float* __restrict__ haug) {
  __shared__ float sH[64][65];
  __shared__ float sO1[64][65];
  __shared__ float sO2[64][65];
  __shared__ float sW2[3*16*64];
  const int g = blockIdx.y;
  const int n0 = blockIdx.x*64;
  const int tid = threadIdx.x;

  for (int idx = tid; idx < 64*64; idx += 256) {
    int i = idx & 63, c = idx >> 6;
    int n = n0 + i;
    float hv = 0.f;
    if (n < NN) {
      size_t jr = (size_t)(g*64 + c)*NPAD + n;
      hv = (float)Hhi[jr] + (float)Hlo[jr];
    }
    sH[i][c] = hv;
  }
  for (int idx = tid; idx < 64*64; idx += 256) {
    int c = idx & 63, i = idx >> 6;
    int n = n0 + i;
    float o1 = 0.f, o2 = 0.f;
    if (n < NN) {
      size_t orow = (size_t)n*NCOL + g*64 + c;
      o1 = O1[orow]; o2 = O2[orow];
    }
    sO1[i][c] = o1; sO2[i][c] = o2;
  }
  __syncthreads();

  const int nl = tid & 63;
  const int dq = tid >> 6;          // 0..3 -> 16 d each
  float acc[16];
  #pragma unroll
  for (int dd = 0; dd < 16; ++dd) acc[dd] = b2[dq*16 + dd];

  for (int cc = 0; cc < 64; cc += 16) {
    __syncthreads();
    for (int idx = tid; idx < 3*16*64; idx += 256) {
      int d = idx & 63, r = idx >> 6;        // r = seg*16 + c
      int seg = r >> 4, c = r & 15;
      sW2[idx] = W2[(seg*64 + cc + c)*64 + d];
    }
    __syncthreads();
    for (int c = 0; c < 16; ++c) {
      float hv = sH[nl][cc + c], o1 = sO1[nl][cc + c], o2 = sO2[nl][cc + c];
      #pragma unroll
      for (int dd = 0; dd < 16; ++dd) {
        int d = dq*16 + dd;
        acc[dd] = fmaf(hv, sW2[(0*16 + c)*64 + d], acc[dd]);
        acc[dd] = fmaf(o1, sW2[(1*16 + c)*64 + d], acc[dd]);
        acc[dd] = fmaf(o2, sW2[(2*16 + c)*64 + d], acc[dd]);
      }
    }
  }

  int n = n0 + nl;
  if (n < NN) {
    #pragma unroll
    for (int dd = 0; dd < 16; ++dd) {
      int d = dq*16 + dd;
      if (g < BT) h2[((size_t)g*NN + n)*64 + d] = acc[dd];
      else        haug[(size_t)n*64 + d] = acc[dd];
    }
  }
}

// ---------------- K6: imputation head y = relu(h2@Wf1+bf1)@Wf2+bf2 ----------------
__global__ __launch_bounds__(128)
void k_head(const float* __restrict__ h2, const float* __restrict__ Wf1,
            const float* __restrict__ bf1, const float* __restrict__ Wf2,
            const float* __restrict__ bf2, float* __restrict__ y) {
  __shared__ float sH[128][65];
  __shared__ float sW1[64*32];
  __shared__ float sW2[32];
  __shared__ float sb1[32];
  const int g = blockIdx.y;
  const int n0 = blockIdx.x*128;
  for (int idx = threadIdx.x; idx < 128*64; idx += 128) {
    int d = idx & 63, i = idx >> 6;
    float v = (n0 + i < NN) ? h2[((size_t)g*NN + n0 + i)*64 + d] : 0.f;
    sH[i][d] = v;
  }
  for (int idx = threadIdx.x; idx < 64*32; idx += 128) sW1[idx] = Wf1[idx];
  if (threadIdx.x < 32) { sW2[threadIdx.x] = Wf2[threadIdx.x]; sb1[threadIdx.x] = bf1[threadIdx.x]; }
  __syncthreads();
  int i = threadIdx.x;
  int n = n0 + i;
  if (n < NN) {
    float out = bf2[0];
    for (int j = 0; j < 32; ++j) {
      float s = sb1[j];
      #pragma unroll 8
      for (int d = 0; d < 64; ++d) s = fmaf(sH[i][d], sW1[d*32 + j], s);
      out = fmaf(fmaxf(s, 0.f), sW2[j], out);
    }
    y[(size_t)g*NN + n] = out;
  }
}

// ---------------- K7: per-item InfoNCE contributions ----------------
__global__ __launch_bounds__(256)
void k_loss_items(const float* __restrict__ h2, const float* __restrict__ haug,
                  const int* __restrict__ missing, const int* __restrict__ posi,
                  const int* __restrict__ negi, float* __restrict__ items) {
  const int wave = threadIdx.x >> 5, lane = threadIdx.x & 31;
  const int item = blockIdx.x*8 + wave;
  if (item >= BB*TT*MMIS) return;
  const int mi = item % MMIS;
  const int g  = item / MMIS;        // b*T + t
  const int t  = g % TT;
  const float* base_o = h2 + (size_t)g*NN*64;
  const bool use_aug = (t < MASKT);

  int na = missing[mi];
  float a0 = base_o[(size_t)na*64 + lane];
  float a1 = base_o[(size_t)na*64 + lane + 32];
  float ss = a0*a0 + a1*a1;
  #pragma unroll
  for (int o = 16; o > 0; o >>= 1) ss += __shfl_xor(ss, o, 32);
  float anorm = fmaxf(sqrtf(ss), 1e-8f);

  auto simfn = [&](int node) -> float {
    const float* bp = use_aug ? (haug + (size_t)node*64) : (base_o + (size_t)node*64);
    float b0 = bp[lane], b1 = bp[lane + 32];
    float dt = a0*b0 + a1*b1;
    float n2 = b0*b0 + b1*b1;
    #pragma unroll
    for (int o = 16; o > 0; o >>= 1) { dt += __shfl_xor(dt, o, 32); n2 += __shfl_xor(n2, o, 32); }
    return dt / (anorm * fmaxf(sqrtf(n2), 1e-8f));
  };

  float l0 = simfn(posi[mi]) * 2.0f;       // /TEMP, TEMP = 0.5
  float s[KNEG];
  for (int k = 0; k < KNEG; ++k) s[k] = simfn(negi[mi*KNEG + k]) * 2.0f;
  float mx = l0;
  for (int k = 0; k < KNEG; ++k) mx = fmaxf(mx, s[k]);
  float se = expf(l0 - mx);
  for (int k = 0; k < KNEG; ++k) se += expf(s[k] - mx);
  float lse = mx + logf(se);
  if (lane == 0) items[item] = lse - l0;
}

// ---------------- K8: deterministic loss reduction ----------------
__global__ void k_loss_reduce(const float* __restrict__ items, float* __restrict__ out_loss) {
  __shared__ float red[256];
  float s = 0.f;
  for (int i = threadIdx.x; i < BB*TT*MMIS; i += 256) s += items[i];
  red[threadIdx.x] = s; __syncthreads();
  for (int o = 128; o > 0; o >>= 1) {
    if (threadIdx.x < o) red[threadIdx.x] += red[threadIdx.x + o];
    __syncthreads();
  }
  if (threadIdx.x == 0) out_loss[0] = red[0] / (float)(BB*TT*MMIS);
}

// ---------------- host orchestration ----------------
extern "C" void kernel_launch(void* const* d_in, const int* in_sizes, int n_in,
                              void* d_out, int out_size, void* d_ws, size_t ws_size,
                              hipStream_t stream) {
  (void)in_sizes; (void)n_in; (void)out_size; (void)ws_size;
  const float* x   = (const float*)d_in[0];
  const float* adj = (const float*)d_in[1];
  const float* Ws  = (const float*)d_in[2];
  const float* bs  = (const float*)d_in[3];
  const float* W1  = (const float*)d_in[4];
  const float* b1  = (const float*)d_in[5];
  const float* W2  = (const float*)d_in[6];
  const float* b2  = (const float*)d_in[7];
  const float* Wf1 = (const float*)d_in[8];
  const float* bf1 = (const float*)d_in[9];
  const float* Wf2 = (const float*)d_in[10];
  const float* bf2 = (const float*)d_in[11];
  const int* missing = (const int*)d_in[12];
  const int* posi    = (const int*)d_in[13];
  const int* negi    = (const int*)d_in[14];
  float* y_out    = (float*)d_out;
  float* loss_out = y_out + (size_t)BB*TT*NN;

  char* p = (char*)d_ws;
  auto carve = [&](size_t bytes) -> void* {
    void* r = (void*)p; p += (bytes + 255) & ~(size_t)255; return r;
  };
  float* dinvr = (float*)carve((size_t)NN*4);
  float* dinvc = (float*)carve((size_t)NN*4);
  float* rs1   = (float*)carve((size_t)NN*4);
  float* rs2   = (float*)carve((size_t)NN*4);
  float* coef  = (float*)carve(384*4);
  float* xr    = (float*)carve((size_t)NN*BT*4);
  float* y1    = (float*)carve((size_t)NN*BT*4);
  float* y2    = (float*)carve((size_t)NN*BT*4);
  float* items = (float*)carve((size_t)BB*TT*MMIS*4);
  __bf16* Hhi  = (__bf16*)carve((size_t)NCOL*NPAD*2);
  __bf16* Hlo  = (__bf16*)carve((size_t)NCOL*NPAD*2);
  __bf16* S1hi = (__bf16*)carve((size_t)MPAD*NPAD*2);
  __bf16* S1lo = (__bf16*)carve((size_t)MPAD*NPAD*2);
  __bf16* S2hi = (__bf16*)carve((size_t)MPAD*NPAD*2);
  __bf16* S2lo = (__bf16*)carve((size_t)MPAD*NPAD*2);
  float* O1    = (float*)carve((size_t)NN*NCOL*4);
  float* O2    = (float*)carve((size_t)NN*NCOL*4);
  float* h2    = (float*)carve((size_t)BT*NN*64*4);
  float* haug  = (float*)carve((size_t)NN*64*4);

  k_coef<<<1, 64, 0, stream>>>(Ws, bs, W1, b1, coef);
  k_rowdeg<<<NN, 256, 0, stream>>>(adj, dinvr, rs1);
  k_coldeg<<<(NN + 255)/256, 256, 0, stream>>>(adj, dinvc, rs2);
  k_xr<<<(BT*NN + 255)/256, 256, 0, stream>>>(x, xr);
  k_y12<<<NN, 96, 0, stream>>>(adj, xr, dinvr, dinvc, y1, y2);
  k_h1<<<dim3((NPAD + 255)/256, NGRP), 256, 0, stream>>>(xr, y1, y2, rs1, rs2, coef, Hhi, Hlo);
  k_s1conv<<<dim3((NPAD + 255)/256, MPAD), 256, 0, stream>>>(adj, dinvr, S1hi, S1lo);
  k_s2conv<<<dim3((NPAD + 63)/64, MPAD/64), 256, 0, stream>>>(adj, dinvc, S2hi, S2lo);
  k_mm<<<dim3(NGRP, MPAD/128), 128, 0, stream>>>(S1hi, S1lo, Hhi, Hlo, O1);
  k_mm<<<dim3(NGRP, MPAD/128), 128, 0, stream>>>(S2hi, S2lo, Hhi, Hlo, O2);
  k_combine<<<dim3((NN + 63)/64, NGRP), 256, 0, stream>>>(O1, O2, Hhi, Hlo, W2, b2, h2, haug);
  k_head<<<dim3((NN + 127)/128, BT), 128, 0, stream>>>(h2, Wf1, bf1, Wf2, bf2, y_out);
  k_loss_items<<<(BB*TT*MMIS)/8, 256, 0, stream>>>(h2, haug, missing, posi, negi, items);
  k_loss_reduce<<<1, 256, 0, stream>>>(items, loss_out);
}
